// MaskedAttentionHead_19516331393397
// MI455X (gfx1250) — matrix operand order repaired and checked by reading, not compiled, not run on hardware
//
#include <hip/hip_runtime.h>
#include <hip/hip_bf16.h>

typedef __attribute__((ext_vector_type(16))) _Float16 v16h;
typedef __attribute__((ext_vector_type(8)))  float    v8f;

#define NB   8
#define NN   1024
#define DD   128

// ---------- WMMA helpers (gfx1250: V_WMMA_F32_16X16X32_F16, wave32) ----------

__device__ __forceinline__ v8f wmma16(v16h a, v16h b, v8f c) {
  // 8 args: (neg_a, A, neg_b, B, c_mod, C, reuse_a, reuse_b)
  return __builtin_amdgcn_wmma_f32_16x16x32_f16(false, a, false, b, (short)0, c, false, false);
}

// A-fragment (16x32 f16, MxK). Lane (l16 = M, hi = lane>=16):
//   halves 0..15 = A[M][kk + hi*16 + j]
// -> two contiguous 16B chunks at p and p+8.
__device__ __forceinline__ v16h load_a_h(const _Float16* p) {
  union { uint4 u[2]; v16h h; } cv;
  cv.u[0] = *reinterpret_cast<const uint4*>(p);
  cv.u[1] = *reinterpret_cast<const uint4*>(p + 8);
  return cv.h;
}
__device__ __forceinline__ v16h load_a_f32c(const float* p) {
  v16h a;
#pragma unroll
  for (int j = 0; j < 8; ++j) { a[j] = (_Float16)p[j]; a[8 + j] = (_Float16)p[8 + j]; }
  return a;
}

// B-fragment (32x16 f16, KxN). Lane (l16 = N, hi):
//   halves 0..15 = B[kk + hi*16 + j][N]; with B supplied "column-major as rows"
//   (i.e. src[N][K] row-major) this is 16 contiguous halves at p.
__device__ __forceinline__ v16h load_b_h(const _Float16* p) {
  union { uint4 u[2]; v16h h; } cv;
  cv.u[0] = *reinterpret_cast<const uint4*>(p);
  cv.u[1] = *reinterpret_cast<const uint4*>(p + 8);
  return cv.h;
}
__device__ __forceinline__ v16h load_b_f32c(const float* p) {
  v16h b;
#pragma unroll
  for (int j = 0; j < 16; ++j) b[j] = (_Float16)p[j];
  return b;
}

__device__ __forceinline__ float wave_sum(float v) {
#pragma unroll
  for (int o = 16; o; o >>= 1) v += __shfl_xor(v, o, 32);
  return v;
}
__device__ __forceinline__ float wave_max(float v) {
#pragma unroll
  for (int o = 16; o; o >>= 1) v = fmaxf(v, __shfl_xor(v, o, 32));
  return v;
}

// ---------- Kernel 1: kbias[n][m] = sum_d keys_pos_enc[n][m][d]  (512MB stream) ----------
__global__ __launch_bounds__(256) void reduce_kpe_kernel(const float* __restrict__ kpe,
                                                         float* __restrict__ kbias) {
  const int lane = threadIdx.x & 31, wave = threadIdx.x >> 5;
  const long row = (long)blockIdx.x * 8 + wave;         // [0, N*N)
  const float4 v4 = *reinterpret_cast<const float4*>(kpe + row * DD + lane * 4);
  float s = wave_sum(v4.x + v4.y + v4.z + v4.w);
  if (lane == 0) kbias[row] = s;
}

// ---------- Kernel 2: vbias[n][d] = sum_m values_pos_enc[n][m][d]  (512MB stream) ----------
__global__ __launch_bounds__(512) void reduce_vpe_kernel(const float* __restrict__ vpe,
                                                         float* __restrict__ vbias) {
  __shared__ float ps[4][DD];
  const int n = blockIdx.x;
  const int d = threadIdx.x & 127, part = threadIdx.x >> 7;
  const float* base = vpe + (long)n * NN * DD + d;
  float acc = 0.f;
  for (int m = part; m < NN; m += 4) acc += base[(long)m * DD];
  ps[part][d] = acc;
  __syncthreads();
  if (part == 0) vbias[(long)n * DD + d] = ps[0][d] + ps[1][d] + ps[2][d] + ps[3][d];
}

// ---------- Kernel 3: v = values @ Wv^T ; emit v_f32 [B*N,D] and vT_h [B][D][N] f16 ----------
__global__ __launch_bounds__(256) void proj_v_kernel(const float* __restrict__ values,
                                                     const float* __restrict__ Wv,
                                                     float* __restrict__ v_f32,
                                                     _Float16* __restrict__ vT_h) {
  const int lane = threadIdx.x & 31, wave = threadIdx.x >> 5;
  const int hi = lane >> 4, l16 = lane & 15;
  const int gid = blockIdx.x * 8 + wave;                 // 4096 tiles
  const int tileM = gid >> 3, tileN = gid & 7;
  const int arow = tileM * 16 + l16;                     // [0,8192)
  const int bcol = tileN * 16 + l16;                     // [0,128)
  v8f c = {};
#pragma unroll
  for (int t = 0; t < 4; ++t) {
    const int kk = t * 32;
    v16h a = load_a_f32c(values + (long)arow * DD + kk + hi * 16);
    v16h b = load_b_f32c(Wv + (long)bcol * DD + kk + hi * 16);   // B(K,N)=Wv[N][K]
    c = wmma16(a, b, c);
  }
#pragma unroll
  for (int r = 0; r < 8; ++r) {
    const int m = tileM * 16 + r + hi * 8;
    const int d = tileN * 16 + l16;
    v_f32[(long)m * DD + d] = c[r];
    const int bIdx = m >> 10, nIdx = m & 1023;
    vT_h[((long)bIdx * DD + d) * NN + nIdx] = (_Float16)c[r];
  }
}

// ---------- Kernel 4: k = v@Wk^T, q = v@Wq^T ; emit f16 row-major [B*N,D] ----------
__global__ __launch_bounds__(256) void proj_kq_kernel(const float* __restrict__ v_f32,
                                                      const float* __restrict__ Wk,
                                                      const float* __restrict__ Wq,
                                                      _Float16* __restrict__ k_h,
                                                      _Float16* __restrict__ q_h) {
  const int lane = threadIdx.x & 31, wave = threadIdx.x >> 5;
  const int hi = lane >> 4, l16 = lane & 15;
  const int gid = blockIdx.x * 8 + wave;
  const int tileM = gid >> 3, tileN = gid & 7;
  const int arow = tileM * 16 + l16;
  const int bcol = tileN * 16 + l16;
  v8f ck = {}, cq = {};
#pragma unroll
  for (int t = 0; t < 4; ++t) {
    const int kk = t * 32;
    v16h a  = load_a_f32c(v_f32 + (long)arow * DD + kk + hi * 16);
    v16h bk = load_b_f32c(Wk + (long)bcol * DD + kk + hi * 16);
    v16h bq = load_b_f32c(Wq + (long)bcol * DD + kk + hi * 16);
    ck = wmma16(a, bk, ck);
    cq = wmma16(a, bq, cq);
  }
#pragma unroll
  for (int r = 0; r < 8; ++r) {
    const int m = tileM * 16 + r + hi * 8;
    const int d = tileN * 16 + l16;
    k_h[(long)m * DD + d] = (_Float16)ck[r];
    q_h[(long)m * DD + d] = (_Float16)cq[r];
  }
}

// ---------- Kernel 5: energy = q.k^T + kbias, mask, softmax(/sqrt(D)) -> attn f16 ----------
// One block = one batch, 16 rows; 8 waves each do 8 column tiles; softmax over full row in LDS.
__global__ __launch_bounds__(256) void energy_softmax_kernel(const _Float16* __restrict__ q_h,
                                                             const _Float16* __restrict__ k_h,
                                                             const float* __restrict__ kbias,
                                                             const int* __restrict__ mask,
                                                             _Float16* __restrict__ attn_h) {
  __shared__ float es[16][NN];                           // 64 KB (LDS is 320KB/WGP)
  const int lane = threadIdx.x & 31, wave = threadIdx.x >> 5;
  const int hi = lane >> 4, l16 = lane & 15;
  const int bIdx = blockIdx.x >> 6, nt = blockIdx.x & 63;
  const int n0 = nt * 16;
  const float scale = 0.08838834764831845f;              // 1/sqrt(128)

  const _Float16* qbase = q_h + ((long)bIdx * NN + n0 + l16) * DD;
  v16h aA[4];
#pragma unroll
  for (int t = 0; t < 4; ++t) aA[t] = load_a_h(qbase + t * 32 + hi * 16);

  for (int ci = 0; ci < 8; ++ci) {
    const int ct = wave + ci * 8;                        // column tile [0,64)
    const int mcol = ct * 16 + l16;
    v8f c = {};
#pragma unroll
    for (int t = 0; t < 4; ++t) {
      v16h b = load_b_h(k_h + ((long)bIdx * NN + mcol) * DD + t * 32 + hi * 16);
      c = wmma16(aA[t], b, c);
    }
#pragma unroll
    for (int r = 0; r < 8; ++r) {
      const int rl = r + hi * 8;
      const int gn = n0 + rl;
      const float e = (c[r] + kbias[(long)gn * NN + mcol]) * scale;
      const int mk = mask[((long)bIdx * NN + gn) * NN + mcol];
      es[rl][mcol] = mk ? e : -__builtin_inff();
    }
  }
  __syncthreads();

  // softmax: each wave owns rows 2w, 2w+1; lane strides columns by 32
  for (int rr = 0; rr < 2; ++rr) {
    const int rl = wave * 2 + rr;
    const int gn = n0 + rl;
    float mx = -__builtin_inff();
#pragma unroll 8
    for (int i = 0; i < 32; ++i) mx = fmaxf(mx, es[rl][lane + i * 32]);
    mx = wave_max(mx);
    float sum = 0.f;
#pragma unroll 8
    for (int i = 0; i < 32; ++i) sum += __expf(es[rl][lane + i * 32] - mx);
    sum = wave_sum(sum);
    const float inv = 1.0f / sum;
    _Float16* arow = attn_h + ((long)bIdx * NN + gn) * NN;
#pragma unroll 8
    for (int i = 0; i < 32; ++i) {
      const int m = lane + i * 32;
      arow[m] = (_Float16)(__expf(es[rl][m] - mx) * inv);
    }
  }
}

// ---------- Kernel 6: out = LN(attn@v + vbias)*gamma + beta + v ----------
// One block = 16 rows; wave w computes d-tile w (8 waves cover D=128); LN fused via LDS.
__global__ __launch_bounds__(256) void attnv_ln_kernel(const _Float16* __restrict__ attn_h,
                                                       const _Float16* __restrict__ vT_h,
                                                       const float* __restrict__ vbias,
                                                       const float* __restrict__ v_f32,
                                                       const float* __restrict__ gamma,
                                                       const float* __restrict__ beta,
                                                       float* __restrict__ out) {
  __shared__ float os[16][DD];                           // 8 KB
  const int lane = threadIdx.x & 31, wave = threadIdx.x >> 5;
  const int hi = lane >> 4, l16 = lane & 15;
  const int bIdx = blockIdx.x >> 6, nt = blockIdx.x & 63;
  const int n0 = nt * 16;
  const int dcol = wave * 16 + l16;

  const _Float16* abase = attn_h + ((long)bIdx * NN + n0 + l16) * NN;
  const _Float16* bbase = vT_h + ((long)bIdx * DD + dcol) * NN;
  v8f c = {};
  for (int t = 0; t < 32; ++t) {                         // K = 1024
    const int kk = t * 32;
    v16h a = load_a_h(abase + kk + hi * 16);
    v16h b = load_b_h(bbase + kk + hi * 16);
    c = wmma16(a, b, c);
  }
#pragma unroll
  for (int r = 0; r < 8; ++r) {
    const int rl = r + hi * 8;
    const int gn = n0 + rl;
    os[rl][dcol] = c[r] + vbias[(long)gn * DD + dcol];
  }
  __syncthreads();

  // layernorm over D=128: wave owns rows 2w, 2w+1; lane owns 4 columns
  for (int rr = 0; rr < 2; ++rr) {
    const int rl = wave * 2 + rr;
    const long grow = (long)bIdx * NN + n0 + rl;
    const float x0 = os[rl][lane * 4 + 0], x1 = os[rl][lane * 4 + 1];
    const float x2 = os[rl][lane * 4 + 2], x3 = os[rl][lane * 4 + 3];
    const float mu = wave_sum(x0 + x1 + x2 + x3) * (1.0f / 128.0f);
    const float d0 = x0 - mu, d1 = x1 - mu, d2 = x2 - mu, d3 = x3 - mu;
    const float var = wave_sum(d0 * d0 + d1 * d1 + d2 * d2 + d3 * d3) * (1.0f / 128.0f);
    const float rs = rsqrtf(var + 1e-5f);
#pragma unroll
    for (int j = 0; j < 4; ++j) {
      const int d = lane * 4 + j;
      const float xd = os[rl][d] - mu;
      out[grow * DD + d] = xd * rs * gamma[d] + beta[d] + v_f32[grow * DD + d];
    }
  }
}

// ---------- launch ----------
extern "C" void kernel_launch(void* const* d_in, const int* in_sizes, int n_in,
                              void* d_out, int out_size, void* d_ws, size_t ws_size,
                              hipStream_t stream) {
  (void)in_sizes; (void)n_in; (void)out_size; (void)ws_size;
  const float* values = (const float*)d_in[0];
  // d_in[1] (keys) and d_in[2] (queries) are unused, faithful to the reference.
  const int*   mask   = (const int*)d_in[3];
  const float* vpe    = (const float*)d_in[4];
  const float* kpe    = (const float*)d_in[5];
  const float* Wv     = (const float*)d_in[6];
  const float* Wk     = (const float*)d_in[7];
  const float* Wq     = (const float*)d_in[8];
  const float* gamma  = (const float*)d_in[9];
  const float* beta   = (const float*)d_in[10];
  float* out = (float*)d_out;

  // workspace layout (30.5 MB total)
  float*    kbias  = (float*)d_ws;                         // 1024*1024 f32  (4 MB)
  float*    vbias  = kbias + (long)NN * NN;                // 1024*128  f32  (0.5 MB)
  float*    v_f32  = vbias + (long)NN * DD;                // 8192*128  f32  (4 MB)
  _Float16* vT_h   = (_Float16*)(v_f32 + (long)NB * NN * DD);  // [B][D][N] f16 (2 MB)
  _Float16* k_h    = vT_h + (long)NB * DD * NN;            // [B*N][D] f16   (2 MB)
  _Float16* q_h    = k_h + (long)NB * NN * DD;             // [B*N][D] f16   (2 MB)
  _Float16* attn_h = q_h + (long)NB * NN * DD;             // [B][N][N] f16  (16 MB)

  reduce_kpe_kernel<<<(NN * NN) / 8, 256, 0, stream>>>(kpe, kbias);
  reduce_vpe_kernel<<<NN, 512, 0, stream>>>(vpe, vbias);
  proj_v_kernel<<<512, 256, 0, stream>>>(values, Wv, v_f32, vT_h);
  proj_kq_kernel<<<512, 256, 0, stream>>>(v_f32, Wk, Wq, k_h, q_h);
  energy_softmax_kernel<<<NB * 64, 256, 0, stream>>>(q_h, k_h, kbias, mask, attn_h);
  attnv_ln_kernel<<<NB * 64, 256, 0, stream>>>(attn_h, vT_h, vbias, v_f32, gamma, beta, out);
}
